// S2PAttn_67147518705774
// MI455X (gfx1250) — matrix operand order repaired and checked by reading, not compiled, not run on hardware
//
#include <hip/hip_runtime.h>
#include <math.h>
#include <stdint.h>

// ---------------------------------------------------------------------------
// S2PAttn for MI455X (gfx1250, wave32, WMMA).
//   B=4, M=8192, N=512, D=512, G=33 (padded to 64), TOP_K=20, 5 Sinkhorn iters.
// Score GEMMs run as bf16 v_wmma_f32_16x16x32_bf16 (f32 accumulate): at
// 23.3 TB/s the feats GEMM (17.2 GFLOP / ~132 MB) is memory-bound only at
// bf16-WMMA rate. Tile staging uses gfx1250 GLOBAL_LOAD_ASYNC_TO_LDS_B128
// (ASYNCcnt) so tiles go HBM->LDS with no VGPR round-trip.
// Sinkhorn column subtraction is deferred/fused; final attn row-normalization
// is algebraically dropped (top-k + weight renorm are row-scale invariant).
// ---------------------------------------------------------------------------

typedef __attribute__((ext_vector_type(16))) __bf16 v16bf;
typedef __attribute__((ext_vector_type(8)))  __bf16 v8bf;
typedef __attribute__((ext_vector_type(8)))  float  v8f;

#define NEG_INF (-3.402823466e38f)

// Async HBM->LDS 16-byte copy (gfx1250; tracked by ASYNCcnt).
// ldsaddr: LDS byte offset (low 32 bits of generic LDS address);
// gaddr  : 64-bit global address in a VGPR pair.
__device__ __forceinline__ void async_ld_b128(unsigned ldsaddr, unsigned long long gaddr) {
  asm volatile("global_load_async_to_lds_b128 %0, %1, off"
               :: "v"(ldsaddr), "v"(gaddr) : "memory");
}
__device__ __forceinline__ void wait_asynccnt0() {
  asm volatile("s_wait_asynccnt 0" ::: "memory");
}

// ---------------- bf16 conversion / padding ----------------
// src [rows][K] f32 -> dst [rows][Kp] bf16 (zero pad K..Kp)
__global__ void cvt_pad_bf16(const float* __restrict__ src, __bf16* __restrict__ dst,
                             long rows, int K, int Kp) {
  long i = (long)blockIdx.x * blockDim.x + threadIdx.x;
  long total = rows * (long)Kp;
  if (i >= total) return;
  long r = i / Kp;
  int  k = (int)(i - r * Kp);
  dst[i] = (k < K) ? (__bf16)src[r * (long)K + k] : (__bf16)0.0f;
}

// src [B][Nn][K] f32 -> dst [B][Kp][Nn] bf16 (transposed, zero pad)
__global__ void cvt_T_bf16(const float* __restrict__ src, __bf16* __restrict__ dst,
                           int B, int Nn, int K, int Kp) {
  long i = (long)blockIdx.x * blockDim.x + threadIdx.x;
  long per_b = (long)Kp * Nn;
  long total = (long)B * per_b;
  if (i >= total) return;
  int b = (int)(i / per_b);
  long rem = i - (long)b * per_b;
  int k = (int)(rem / Nn);
  int n = (int)(rem - (long)k * Nn);
  dst[i] = (k < K) ? (__bf16)src[((long)b * Nn + n) * K + k] : (__bf16)0.0f;
}

// ---------------- bf16 WMMA GEMM:  S = scale * X @ NT ----------------
// X  : [B][M][Kp]  bf16 row-major
// NT : [B][Kp][N]  bf16 (node matrix pre-transposed)
// S  : [B][M][N]   f32
// Block tile 128(M) x 64(N), BK=32, 256 threads = 8 waves.
// Wave w owns M-rows [16w,16w+16) x all four 16-wide N tiles.
__global__ void __launch_bounds__(256)
gemm_bf16(const __bf16* __restrict__ X, const __bf16* __restrict__ NT,
          float* __restrict__ S, int M, int N, int Kp, float scale) {
  __shared__ __align__(64) __bf16 lA[128][32];
  __shared__ __align__(64) __bf16 lB[32][64];

  int b  = blockIdx.z;
  int m0 = blockIdx.y * 128;
  int n0 = blockIdx.x * 64;
  int t  = threadIdx.x;
  int w  = t >> 5, lane = t & 31;

  const __bf16* Xb = X  + ((size_t)b * M + m0) * Kp;
  const __bf16* Nb = NT + ((size_t)b * Kp) * N + n0;

  v8f zero = {0.f,0.f,0.f,0.f,0.f,0.f,0.f,0.f};
  v8f acc[4];
#pragma unroll
  for (int i = 0; i < 4; ++i) acc[i] = zero;

  int arow = t >> 1, ahalf = (t & 1) * 16;   // A: each thread 32B (16 bf16)
  int bk   = t >> 3, bchunk = (t & 7) * 8;   // B: each thread 16B (8 bf16)

  // LDS byte offsets for this thread's staging slots (generic LDS address
  // low 32 bits == LDS byte offset per the flat-aperture rules)
  unsigned ldsA = (unsigned)(uintptr_t)&lA[arow][ahalf];
  unsigned ldsB = (unsigned)(uintptr_t)&lB[bk][bchunk];

  for (int k0 = 0; k0 < Kp; k0 += 32) {
    __syncthreads();
    // stage A tile 128x32 and B tile 32x64 directly into LDS (ASYNCcnt path)
    {
      unsigned long long gA =
          (unsigned long long)(uintptr_t)(Xb + (size_t)arow * Kp + k0 + ahalf);
      async_ld_b128(ldsA,       gA);
      async_ld_b128(ldsA + 16u, gA + 16ull);
      unsigned long long gB =
          (unsigned long long)(uintptr_t)(Nb + (size_t)(k0 + bk) * N + bchunk);
      async_ld_b128(ldsB, gB);
    }
    if (k0 + 32 < Kp) { // gfx1250 global_prefetch_b8 for next K tile
      __builtin_prefetch(Xb + (size_t)arow * Kp + k0 + 32 + ahalf, 0, 0);
      __builtin_prefetch(Nb + (size_t)(k0 + 32 + bk) * N + bchunk, 0, 0);
    }
    wait_asynccnt0();
    __syncthreads();

    // A fragment (16x32 bf16): lanes 0-15 hold M=lane, K {0..7,16..23};
    // lanes 16-31 hold M=lane-16, K {8..15,24..31}.
    int ar = (w << 4) + (lane & 15);
    int ak = (lane < 16) ? 0 : 8;
    v8bf alo = *(const v8bf*)&lA[ar][ak];
    v8bf ahi = *(const v8bf*)&lA[ar][ak + 16];
    v16bf a = __builtin_shufflevector(alo, ahi,
        0,1,2,3,4,5,6,7,8,9,10,11,12,13,14,15);

#pragma unroll
    for (int nt = 0; nt < 4; ++nt) {
      // B fragment (32x16 bf16): lane holds N=lane&15; lanes 0-15 K {0..7,16..23}, lanes 16-31 K {8..15,24..31}.
      v16bf bf;
#pragma unroll
      for (int s = 0; s < 16; ++s)
        bf[s] = lB[ak + (s & 7) + ((s >> 3) << 4)][nt * 16 + (lane & 15)];
      acc[nt] = __builtin_amdgcn_wmma_f32_16x16x32_bf16(
          false, a, false, bf, (short)0, acc[nt], false, false);
    }
  }

  // C/D layout: VGPR r, lanes 0-15 -> M=r, lanes 16-31 -> M=8+r; N=lane&15.
  size_t mrow = (size_t)b * M + m0 + (w << 4) + ((lane >> 4) * 8);
#pragma unroll
  for (int nt = 0; nt < 4; ++nt) {
    int n = n0 + nt * 16 + (lane & 15);
#pragma unroll
    for (int r = 0; r < 8; ++r)
      S[(mrow + r) * (size_t)N + n] = acc[nt][r] * scale;
  }
}

// ---------------- Sinkhorn (log domain), batched over Bt=2*B matrices -------
// Row pass: subtract previous column-LSE (if any), then row logsumexp. One
// wave per row (N=512 -> 16 values/lane, kept in registers).
__global__ void __launch_bounds__(256)
row_norm(float* __restrict__ S, const float* __restrict__ clse, int M, int N) {
  int w = threadIdx.x >> 5, lane = threadIdx.x & 31;
  long row = (long)blockIdx.x * 8 + w;
  int bt = (int)(row / M);
  const float* cl = clse ? (clse + (size_t)bt * N) : nullptr;

  float x[16];
#pragma unroll
  for (int j = 0; j < 16; ++j) {
    int n = lane + 32 * j;
    float v = S[row * (long)N + n];
    if (cl) v -= cl[n];
    x[j] = v;
  }
  float mx = NEG_INF;
#pragma unroll
  for (int j = 0; j < 16; ++j) mx = fmaxf(mx, x[j]);
  for (int off = 16; off > 0; off >>= 1) mx = fmaxf(mx, __shfl_xor(mx, off));
  float s = 0.f;
#pragma unroll
  for (int j = 0; j < 16; ++j) s += __expf(x[j] - mx);
  for (int off = 16; off > 0; off >>= 1) s += __shfl_xor(s, off);
  float lse = mx + __logf(s);
#pragma unroll
  for (int j = 0; j < 16; ++j)
    S[row * (long)N + lane + 32 * j] = x[j] - lse;
}

// Column pass 1: streaming LSE over a 256-row chunk per block; thread = column.
__global__ void col_partial(const float* __restrict__ S, float* __restrict__ pmax,
                            float* __restrict__ psum, int M, int N) {
  int n = threadIdx.x, blk = blockIdx.x, bt = blockIdx.y;
  int rows = M / gridDim.x;
  const float* base = S + ((size_t)bt * M + (size_t)blk * rows) * N + n;
  float mx = NEG_INF, sm = 0.f;
  for (int m = 0; m < rows; ++m) {
    float v = base[(size_t)m * N];
    if (v > mx) { sm = sm * __expf(mx - v) + 1.f; mx = v; }
    else         sm += __expf(v - mx);
  }
  size_t o = ((size_t)bt * gridDim.x + blk) * N + n;
  pmax[o] = mx; psum[o] = sm;
}

// Column pass 2: combine chunk partials into col_lse[bt][n].
__global__ void col_combine(const float* __restrict__ pmax, const float* __restrict__ psum,
                            float* __restrict__ clse, int N, int nblk) {
  int n = threadIdx.x, bt = blockIdx.x;
  float gm = NEG_INF, s = 0.f;
  for (int k = 0; k < nblk; ++k) {
    size_t o = ((size_t)bt * nblk + k) * N + n;
    float m_ = pmax[o], s_ = psum[o];
    if (m_ > gm) { s = s * __expf(gm - m_) + s_; gm = m_; }
    else          s += s_ * __expf(m_ - gm);
  }
  clse[(size_t)bt * N + n] = gm + __logf(s);
}

// ---------------- top-k + prototype + output ----------------
// One wave per (b,m) row. s[n] = gs - clse_g + fs - clse_f (row-scale-free).
__global__ void __launch_bounds__(256)
topk_out(const float* __restrict__ Sg, const float* __restrict__ Sf,
         const float* __restrict__ clse,
         const float* __restrict__ node_feats, const float* __restrict__ node_fpfhs,
         const float* __restrict__ xyz_feats, const float* __restrict__ xyz_fpfhs,
         float* __restrict__ out, int B, int M, int Nn, int D, int G) {
  int w = threadIdx.x >> 5, lane = threadIdx.x & 31;
  long row = (long)blockIdx.x * 8 + w;          // [0, B*M)
  int b = (int)(row / M);
  const float* clg = clse + (size_t)b * Nn;
  const float* clf = clse + (size_t)(B + b) * Nn;

  float x[16];
#pragma unroll
  for (int j = 0; j < 16; ++j) {
    int n = lane + 32 * j;
    x[j] = Sg[row * (long)Nn + n] - clg[n] + Sf[row * (long)Nn + n] - clf[n];
  }

  float vals[20]; int idxs[20];
#pragma unroll
  for (int r = 0; r < 20; ++r) {
    float best = NEG_INF; int bi = 0;
#pragma unroll
    for (int j = 0; j < 16; ++j)
      if (x[j] > best) { best = x[j]; bi = lane + 32 * j; }
    for (int off = 16; off > 0; off >>= 1) {
      float ov = __shfl_xor(best, off);
      int   oi = __shfl_xor(bi, off);
      if (ov > best) { best = ov; bi = oi; }
    }
    vals[r] = best; idxs[r] = bi;
#pragma unroll
    for (int j = 0; j < 16; ++j)
      if (bi == lane + 32 * j) x[j] = NEG_INF;   // owner clears winner
  }

  // softmax-renormalize the top-20 weights (vals[0] is the max)
  float wts[20], wsum = 0.f, mx = vals[0];
#pragma unroll
  for (int r = 0; r < 20; ++r) { wts[r] = __expf(vals[r] - mx); wsum += wts[r]; }
  float inv = 1.0f / wsum;
#pragma unroll
  for (int r = 0; r < 20; ++r) wts[r] *= inv;

  // prototype feats: [D=512] -> 16 per lane, gathers coalesced across lanes
  const float* nf = node_feats + (size_t)b * Nn * D;
  size_t orow = (size_t)row * D;
  for (int jd = 0; jd < 16; ++jd) {
    int d = lane + 32 * jd;
    float acc = 0.f;
#pragma unroll
    for (int r = 0; r < 20; ++r) acc += wts[r] * nf[(size_t)idxs[r] * D + d];
    out[orow + d] = 0.5f * (acc + xyz_feats[orow + d]);
  }
  // prototype fpfhs: [G=33]
  const float* ng = node_fpfhs + (size_t)b * Nn * G;
  size_t feats_total = (size_t)B * M * D;
  size_t grow = (size_t)row * G;
  for (int d = lane; d < G; d += 32) {
    float acc = 0.f;
#pragma unroll
    for (int r = 0; r < 20; ++r) acc += wts[r] * ng[(size_t)idxs[r] * G + d];
    out[feats_total + grow + d] = 0.5f * (acc + xyz_fpfhs[grow + d]);
  }
}

// ---------------------------------------------------------------------------
extern "C" void kernel_launch(void* const* d_in, const int* in_sizes, int n_in,
                              void* d_out, int out_size, void* d_ws, size_t ws_size,
                              hipStream_t stream) {
  const float* node_feats = (const float*)d_in[2];
  const float* xyz_feats  = (const float*)d_in[3];
  const float* node_fpfhs = (const float*)d_in[4];
  const float* xyz_fpfhs  = (const float*)d_in[5];
  float* out = (float*)d_out;

  const int B = 4, M = 8192, Nn = 512, D = 512, G = 33, Gp = 64;
  const int Bt = 2 * B;            // g- and f-matrices batched together
  const int CBLK = 32;             // column-pass chunks (256 rows each)

  // carve workspace (256B aligned)
  char* p = (char*)d_ws;
  auto carve = [&](size_t bytes) {
    char* q = p; p += (bytes + 255) & ~(size_t)255; return q;
  };
  __bf16* Xf  = (__bf16*)carve((size_t)B * M * D  * 2);   // xyz_feats  bf16
  __bf16* NfT = (__bf16*)carve((size_t)B * D * Nn * 2);   // node_feats^T bf16
  __bf16* Xg  = (__bf16*)carve((size_t)B * M * Gp * 2);   // xyz_fpfhs  bf16 (pad 33->64)
  __bf16* NgT = (__bf16*)carve((size_t)B * Gp * Nn * 2);  // node_fpfhs^T bf16
  float*  S   = (float*)carve((size_t)Bt * M * Nn * 4);   // [g(4) | f(4)] log-scores
  float*  Sg  = S;
  float*  Sf  = S + (size_t)B * M * Nn;
  float* pmax = (float*)carve((size_t)Bt * CBLK * Nn * 4);
  float* psum = (float*)carve((size_t)Bt * CBLK * Nn * 4);
  float* clse = (float*)carve((size_t)Bt * Nn * 4);

  // 1) precision conversion / layout
  {
    long tf = (long)B * M * D;
    cvt_pad_bf16<<<(unsigned)((tf + 255) / 256), 256, 0, stream>>>(xyz_feats, Xf, (long)B * M, D, D);
    long tn = (long)B * D * Nn;
    cvt_T_bf16<<<(unsigned)((tn + 255) / 256), 256, 0, stream>>>(node_feats, NfT, B, Nn, D, D);
    long tg = (long)B * M * Gp;
    cvt_pad_bf16<<<(unsigned)((tg + 255) / 256), 256, 0, stream>>>(xyz_fpfhs, Xg, (long)B * M, G, Gp);
    long tgn = (long)B * Gp * Nn;
    cvt_T_bf16<<<(unsigned)((tgn + 255) / 256), 256, 0, stream>>>(node_fpfhs, NgT, B, Nn, G, Gp);
  }

  // 2) WMMA score GEMMs (scale = 1/sqrt(K_true))
  dim3 gg(Nn / 64, M / 128, B);
  gemm_bf16<<<gg, 256, 0, stream>>>(Xg, NgT, Sg, M, Nn, Gp, 1.0f / sqrtf((float)G));
  gemm_bf16<<<gg, 256, 0, stream>>>(Xf, NfT, Sf, M, Nn, D,  1.0f / sqrtf((float)D));

  // 3) Sinkhorn, 5 iterations, both matrices batched (col-subtract fused
  //    into next row pass; final col-LSE left pending for the output kernel)
  for (int it = 0; it < 5; ++it) {
    row_norm<<<Bt * M / 8, 256, 0, stream>>>(S, it ? clse : nullptr, M, Nn);
    col_partial<<<dim3(CBLK, Bt), Nn, 0, stream>>>(S, pmax, psum, M, Nn);
    col_combine<<<Bt, Nn, 0, stream>>>(pmax, psum, clse, Nn, CBLK);
  }

  // 4) fused top-20 + weight renorm + prototype gather + residual + /2
  topk_out<<<B * M / 8, 256, 0, stream>>>(Sg, Sf, clse, node_feats, node_fpfhs,
                                          xyz_feats, xyz_fpfhs, out, B, M, Nn, D, G);
}
